// MuNet_43928925503537
// MI455X (gfx1250) — matrix operand order repaired and checked by reading, not compiled, not run on hardware
//
#include <hip/hip_runtime.h>
#include <math.h>

typedef __bf16 bf16_t;
typedef __attribute__((ext_vector_type(16))) __bf16 v16bf;
typedef __attribute__((ext_vector_type(8)))  __bf16 v8bf;
typedef __attribute__((ext_vector_type(8)))  float  v8f;
typedef __attribute__((ext_vector_type(4)))  float  v4f;

#define D_IN  360
#define K1P   384          // D_IN padded to multiple of 32
#define H1    128
#define H2    128
#define H3    64
#define WAVES_PER_WG  4
#define NTHREADS      (WAVES_PER_WG * 32)              // 128
#define ROWS_PER_WAVE 16
#define ROWS_PER_WG   (WAVES_PER_WG * ROWS_PER_WAVE)   // 64
#define TILE_LINES    ((ROWS_PER_WG * D_IN * 4) / 128) // 720 x 128B lines per tile
#define MAX_WGS       1024

static __device__ __forceinline__ v16bf cat8(v8bf lo, v8bf hi) {
  v16bf r;
#pragma unroll
  for (int i = 0; i < 8; ++i) { r[i] = lo[i]; r[i + 8] = hi[i]; }
  return r;
}

static __device__ __forceinline__ v8bf lds8(const bf16_t* p) {
  return *(const v8bf*)__builtin_assume_aligned(p, 16);
}

static __device__ __forceinline__ v8f splat8(float v) {
  v8f r;
#pragma unroll
  for (int i = 0; i < 8; ++i) r[i] = v;
  return r;
}

__global__ __launch_bounds__(NTHREADS, 2)
void munet_fused_kernel(const float* __restrict__ x,
                        const float* __restrict__ W1, const float* __restrict__ b1,
                        const float* __restrict__ W2, const float* __restrict__ b2,
                        const float* __restrict__ W3, const float* __restrict__ b3,
                        const float* __restrict__ Wmu, const float* __restrict__ bmu,
                        float* __restrict__ out, int numTiles)
{
  // Transposed [n][k] bf16 weight tiles: B-fragments are contiguous 16B LDS loads.
  __shared__ bf16_t sW1t[H1 * K1P];                          // 96 KB
  __shared__ bf16_t sW2t[H2 * H1];                           // 32 KB
  __shared__ bf16_t sW3t[H3 * H2];                           // 16 KB
  __shared__ bf16_t hbuf[WAVES_PER_WG][ROWS_PER_WAVE * H1];  // 16 KB
  __shared__ float  sWmu[H3 * 2];                            // 512 B

  const int tid    = threadIdx.x;
  const int wave   = tid >> 5;
  const int lane   = tid & 31;
  const int mrow   = lane & 15;        // row within M-tile / column within N-tile
  const int hihalf = (lane >> 4) & 1;  // lanes 16..31

  // ------------- stage weights once per (persistent) workgroup -------------
  for (int idx = tid; idx < H1 * K1P; idx += NTHREADS) {
    int n = idx / K1P;
    int k = idx - n * K1P;
    sW1t[idx] = (k < D_IN) ? (bf16_t)W1[k * H1 + n] : (bf16_t)0.0f;
  }
  for (int idx = tid; idx < H2 * H1; idx += NTHREADS) {
    int n = idx >> 7, k = idx & (H1 - 1);
    sW2t[idx] = (bf16_t)W2[k * H2 + n];
  }
  for (int idx = tid; idx < H3 * H2; idx += NTHREADS) {
    int n = idx >> 7, k = idx & (H2 - 1);
    sW3t[idx] = (bf16_t)W3[k * H3 + n];
  }
  if (tid < H3 * 2) sWmu[tid] = Wmu[tid];
  __syncthreads();

  // ------------- per-lane invariants hoisted out of the tile loop -------------
  float bias1[8], bias2[8], bias3[4];
#pragma unroll
  for (int nt = 0; nt < 8; ++nt) bias1[nt] = b1[nt * 16 + mrow];
#pragma unroll
  for (int nt = 0; nt < 8; ++nt) bias2[nt] = b2[nt * 16 + mrow];
#pragma unroll
  for (int nt = 0; nt < 4; ++nt) bias3[nt] = b3[nt * 16 + mrow];
  const int   mcol = hihalf;            // head: lane -> (row=mrow, col=mcol)
  const float bm   = bmu[mcol];
  bf16_t* hb = hbuf[wave];

  // ===================== persistent tile loop =====================
  for (int tile = blockIdx.x; tile < numTiles; tile += gridDim.x) {
    // Prefetch next tile's x stream (global_prefetch_b8) to overlap HBM latency
    // with this tile's 144 WMMAs.
    const int nxt = tile + gridDim.x;
    if (nxt < numTiles) {
      const float* nx = x + (size_t)nxt * ROWS_PER_WG * D_IN;
#pragma unroll
      for (int p = 0; p < 6; ++p) {
        int line = tid + p * NTHREADS;
        if (line < TILE_LINES) __builtin_prefetch(nx + line * 32, 0, 0);
      }
    }

    const int rowBase = tile * ROWS_PER_WG + wave * ROWS_PER_WAVE;
    const float* xrow = x + (size_t)(rowBase + mrow) * D_IN;

    // ---------------- Layer 1: [16,360] @ [360,128] ----------------
    v8f acc1[8];
#pragma unroll
    for (int nt = 0; nt < 8; ++nt) acc1[nt] = splat8(bias1[nt]);

#pragma unroll
    for (int c = 0; c < 12; ++c) {
      // A-fragment (16x32 bf16): lo run at k0, hi run at k0+16 (ISA A layout)
      v16bf a;
      const int k0 = c * 32 + hihalf * 8;
      if (c < 11) {
        v4f x0 = *(const v4f*)(xrow + k0);
        v4f x1 = *(const v4f*)(xrow + k0 + 4);
        v4f x2 = *(const v4f*)(xrow + k0 + 16);
        v4f x3 = *(const v4f*)(xrow + k0 + 20);
#pragma unroll
        for (int i = 0; i < 4; ++i) {
          a[i]      = (bf16_t)x0[i];
          a[4 + i]  = (bf16_t)x1[i];
          a[8 + i]  = (bf16_t)x2[i];
          a[12 + i] = (bf16_t)x3[i];
        }
      } else {  // tail chunk: K 352..383, zero-pad past 359
#pragma unroll
        for (int e = 0; e < 16; ++e) {
          int k = k0 + ((e < 8) ? e : (8 + e));
          a[e] = (k < D_IN) ? (bf16_t)xrow[k] : (bf16_t)0.0f;
        }
      }
#pragma unroll
      for (int nt = 0; nt < 8; ++nt) {
        const bf16_t* bp = &sW1t[(nt * 16 + mrow) * K1P + c * 32 + hihalf * 8];
        v16bf b = cat8(lds8(bp), lds8(bp + 16));
        acc1[nt] = __builtin_amdgcn_wmma_f32_16x16x32_bf16(
            false, a, false, b, (short)0, acc1[nt], false, false);
      }
    }
    // ReLU + h1 -> LDS ([16][128] bf16, C/D layout -> row-major)
#pragma unroll
    for (int nt = 0; nt < 8; ++nt) {
#pragma unroll
      for (int r = 0; r < 8; ++r) {
        int m = r + hihalf * 8;
        hb[m * H1 + nt * 16 + mrow] = (bf16_t)fmaxf(acc1[nt][r], 0.0f);
      }
    }

    // ---------------- Layer 2: [16,128] @ [128,128] ----------------
    v8f acc2[8];
#pragma unroll
    for (int nt = 0; nt < 8; ++nt) acc2[nt] = splat8(bias2[nt]);

#pragma unroll
    for (int c = 0; c < 4; ++c) {
      const bf16_t* ap = &hb[mrow * H1 + c * 32 + hihalf * 8];
      v16bf a = cat8(lds8(ap), lds8(ap + 16));
#pragma unroll
      for (int nt = 0; nt < 8; ++nt) {
        const bf16_t* bp = &sW2t[(nt * 16 + mrow) * H1 + c * 32 + hihalf * 8];
        v16bf b = cat8(lds8(bp), lds8(bp + 16));
        acc2[nt] = __builtin_amdgcn_wmma_f32_16x16x32_bf16(
            false, a, false, b, (short)0, acc2[nt], false, false);
      }
    }
#pragma unroll
    for (int nt = 0; nt < 8; ++nt) {
#pragma unroll
      for (int r = 0; r < 8; ++r) {
        int m = r + hihalf * 8;
        hb[m * H1 + nt * 16 + mrow] = (bf16_t)fmaxf(acc2[nt][r], 0.0f);
      }
    }

    // ---------------- Layer 3: [16,128] @ [128,64] ----------------
    v8f acc3[4];
#pragma unroll
    for (int nt = 0; nt < 4; ++nt) acc3[nt] = splat8(bias3[nt]);

#pragma unroll
    for (int c = 0; c < 4; ++c) {
      const bf16_t* ap = &hb[mrow * H1 + c * 32 + hihalf * 8];
      v16bf a = cat8(lds8(ap), lds8(ap + 16));
#pragma unroll
      for (int nt = 0; nt < 4; ++nt) {
        const bf16_t* bp = &sW3t[(nt * 16 + mrow) * H2 + c * 32 + hihalf * 8];
        v16bf b = cat8(lds8(bp), lds8(bp + 16));
        acc3[nt] = __builtin_amdgcn_wmma_f32_16x16x32_bf16(
            false, a, false, b, (short)0, acc3[nt], false, false);
      }
    }
    // h3 -> LDS as [16][64] bf16 (reuses hbuf; all h2 reads are complete)
#pragma unroll
    for (int nt = 0; nt < 4; ++nt) {
#pragma unroll
      for (int r = 0; r < 8; ++r) {
        int m = r + hihalf * 8;
        hb[m * H3 + nt * 16 + mrow] = (bf16_t)fmaxf(acc3[nt][r], 0.0f);
      }
    }

    // ---------------- Head: [16,64] @ [64,2] + relu / 3*tanh ----------------
    float accm = 0.0f;
#pragma unroll
    for (int k = 0; k < H3; ++k)
      accm += (float)hb[mrow * H3 + k] * sWmu[k * 2 + mcol];
    accm += bm;
    float res = (mcol == 0) ? fmaxf(accm, 0.0f) : 3.0f * tanhf(accm);
    out[(size_t)(rowBase + mrow) * 2 + mcol] = res;
  }
}

extern "C" void kernel_launch(void* const* d_in, const int* in_sizes, int n_in,
                              void* d_out, int out_size, void* d_ws, size_t ws_size,
                              hipStream_t stream) {
  const float* x   = (const float*)d_in[0];
  const float* W1  = (const float*)d_in[1];
  const float* b1  = (const float*)d_in[2];
  const float* W2  = (const float*)d_in[3];
  const float* b2  = (const float*)d_in[4];
  const float* W3  = (const float*)d_in[5];
  const float* b3  = (const float*)d_in[6];
  const float* Wmu = (const float*)d_in[7];
  const float* bmu = (const float*)d_in[8];
  float* out = (float*)d_out;

  const int batch    = in_sizes[0] / D_IN;        // 262144
  const int numTiles = batch / ROWS_PER_WG;       // 4096 (divides exactly)
  const int grid     = (numTiles < MAX_WGS) ? numTiles : MAX_WGS;  // persistent WGs
  munet_fused_kernel<<<grid, NTHREADS, 0, stream>>>(
      x, W1, b1, W2, b2, W3, b3, Wmu, bmu, out, numTiles);
}
